// OCMod_13932873908296
// MI455X (gfx1250) — matrix operand order repaired and checked by reading, not compiled, not run on hardware
//
#include <hip/hip_runtime.h>
#include <hip/hip_bf16.h>
#include <math.h>

// ---------------------------------------------------------------------------
// MoE per-species MLP (256 -> 128 -> 64 -> 1, erf-GELU) for MI455X (gfx1250).
// Strategy: bin atoms by species into 16-aligned index regions, then run
// species-uniform 16-atom WMMA tiles (v_wmma_f32_16x16x32_f16) with the
// expert's weights pre-converted to f16 WMMA B-fragment layout and staged in
// LDS. Streams oc_density exactly once (~1 GB HBM -> memory-bound optimum).
// ---------------------------------------------------------------------------

typedef __attribute__((ext_vector_type(16))) _Float16 v16h;
typedef __attribute__((ext_vector_type(8)))  _Float16 v8h;
typedef __attribute__((ext_vector_type(8)))  float    v8f;

#define NFEAT 256
#define NH1   128
#define NH2   64
#define NTYPE 4
#define WAVES 8            // waves per block in main kernel (256 threads)
#define BPS   256          // blocks per species -> grid = 4*BPS

// LDS layout (bytes), dynamic shared = 128 KB
#define LDS_FRAG0 0        // 8 n-tiles * 8 k-chunks * 1KB = 64 KB
#define LDS_FRAG1 65536    // 4 * 4 * 1KB = 16 KB
#define LDS_H1    81920    // 8 waves * 4 KB (16x128 f16)
#define LDS_H2    114688   // 8 waves * 2 KB (16x64 f16)
#define LDS_TOTAL 131072

__device__ __forceinline__ float gelu_erf(float x) {
  return 0.5f * x * (1.0f + erff(x * 0.70710678118654752f));
}

// --- binning kernels --------------------------------------------------------

__global__ void k_init(int* __restrict__ hdr, int* __restrict__ idxbuf, int total) {
  int i = blockIdx.x * blockDim.x + threadIdx.x;
  if (i < 16) hdr[i] = 0;                 // counts[0..3], starts[4..8], cursors[12..15]
  if (i < total) idxbuf[i] = -1;
}

__global__ void k_count(const int* __restrict__ species, int n, int* __restrict__ counts) {
  int i = blockIdx.x * blockDim.x + threadIdx.x;
  if (i < n) {
    int s = species[i] & 3;
    atomicAdd(&counts[s], 1);
  }
}

__global__ void k_starts(const int* __restrict__ counts, int* __restrict__ starts) {
  // single thread: starts[t+1] = roundup16(starts[t] + counts[t])
  int p = 0;
  starts[0] = 0;
  for (int t = 0; t < NTYPE; ++t) {
    p += counts[t];
    p = (p + 15) & ~15;
    starts[t + 1] = p;
  }
}

__global__ void k_scatter(const int* __restrict__ species, int n,
                          const int* __restrict__ starts, int* __restrict__ cursors,
                          int* __restrict__ idxbuf) {
  int i = blockIdx.x * blockDim.x + threadIdx.x;
  if (i < n) {
    int s = species[i] & 3;
    int pos = starts[s] + atomicAdd(&cursors[s], 1);
    idxbuf[pos] = i;
  }
}

// --- weight -> f16 WMMA B-fragment layout -----------------------------------
// B fragment (32x16, f16): lane l holds column N = l%16; K = kbase + (l<16?0:8) + (e<8?e:e+8),
// e = 0..15. Stored as frag[fragId][lane][e] contiguous (1 KB per fragment).

__global__ void k_prep0(const float* __restrict__ W0, _Float16* __restrict__ frag0) {
  int tid = blockIdx.x * 256 + threadIdx.x;       // 0 .. 4*8*8*512-1
  int f = tid >> 9; int r = tid & 511; int lane = r >> 4; int e = r & 15;
  int t = f >> 6; int n = (f >> 3) & 7; int kc = f & 7;
  int col = n * 16 + (lane & 15);                 // H1 output index
  int k   = kc * 32 + ((lane < 16) ? 0 : 8) + ((e < 8) ? e : e + 8); // feature index
  frag0[tid] = (_Float16)W0[(t * NH1 + col) * NFEAT + k];
}

__global__ void k_prep1(const float* __restrict__ W1, _Float16* __restrict__ frag1) {
  int tid = blockIdx.x * 256 + threadIdx.x;       // 0 .. 4*4*4*512-1
  int f = tid >> 9; int r = tid & 511; int lane = r >> 4; int e = r & 15;
  int t = f >> 4; int n = (f >> 2) & 3; int kc = f & 3;
  int col = n * 16 + (lane & 15);                 // H2 output index
  int k   = kc * 32 + ((lane < 16) ? 0 : 8) + ((e < 8) ? e : e + 8); // H1 index
  frag1[tid] = (_Float16)W1[(t * NH2 + col) * NH1 + k];
}

// --- main MoE MLP kernel ----------------------------------------------------

__global__ void __launch_bounds__(256)
k_moe_mlp(const float* __restrict__ X, const int* __restrict__ idxbuf,
          const int* __restrict__ starts,
          const _Float16* __restrict__ frag0g, const _Float16* __restrict__ frag1g,
          const float* __restrict__ B0, const float* __restrict__ B1,
          const float* __restrict__ W2, const float* __restrict__ B2,
          float* __restrict__ out) {
  extern __shared__ char smem[];
  const int s    = blockIdx.x & 3;      // species this block serves
  const int bls  = blockIdx.x >> 2;
  const int bps  = gridDim.x >> 2;
  const int lane = threadIdx.x & 31;
  const int wave = threadIdx.x >> 5;
  const int colL = lane & 15;
  const int hiH  = (lane < 16) ? 0 : 1;

  // Stage this species' weight fragments into LDS (read-only afterwards).
  {
    const uint4* s0 = (const uint4*)(frag0g + (size_t)s * (8 * 8 * 512));
    uint4* d0 = (uint4*)(smem + LDS_FRAG0);
    for (int i = threadIdx.x; i < 4096; i += 256) d0[i] = s0[i];
    const uint4* s1 = (const uint4*)(frag1g + (size_t)s * (4 * 4 * 512));
    uint4* d1 = (uint4*)(smem + LDS_FRAG1);
    for (int i = threadIdx.x; i < 1024; i += 256) d1[i] = s1[i];
  }
  __syncthreads();

  _Float16* h1ws = (_Float16*)(smem + LDS_H1 + wave * 4096);
  _Float16* h2ws = (_Float16*)(smem + LDS_H2 + wave * 2048);

  // Per-lane constants hoisted out of the group loop.
  float bias0[8], bias1[4], w2r[32];
#pragma unroll
  for (int n = 0; n < 8; ++n) bias0[n] = B0[s * NH1 + n * 16 + colL];
#pragma unroll
  for (int n = 0; n < 4; ++n) bias1[n] = B1[s * NH2 + n * 16 + colL];
#pragma unroll
  for (int j = 0; j < 32; ++j) w2r[j] = W2[s * NH2 + hiH * 32 + j];
  const float b2s = B2[s];

  const int st = starts[s];
  const int en = starts[s + 1];
  const int ngroups = (en - st) >> 4;
  const int aoff = hiH * 8;
  const int M0   = hiH * 8;
  const v8f vzero = {0.f, 0.f, 0.f, 0.f, 0.f, 0.f, 0.f, 0.f};

  for (int g = bls * WAVES + wave; g < ngroups; g += bps * WAVES) {
    const int slot0 = st + (g << 4);
    const int myidx = idxbuf[slot0 + colL];          // row this lane represents
    const size_t row = (size_t)(myidx < 0 ? 0 : myidx);
    const float* xrow = X + row * NFEAT;

    // ---- Layer 0: [16 x 256] x [256 x 128] -> [16 x 128] -----------------
    v8f acc[8];
#pragma unroll
    for (int n = 0; n < 8; ++n) acc[n] = vzero;

#pragma unroll
    for (int kc = 0; kc < 8; ++kc) {
      // Build A fragment (16x32 f16): lane l = row l%16,
      // K = kc*32 + (e<8 ? e : e+8) + (l<16 ? 0 : 8).
      const float* p = xrow + kc * 32 + aoff;
      float4 r0 = *(const float4*)(p);
      float4 r1 = *(const float4*)(p + 4);
      float4 r2 = *(const float4*)(p + 16);
      float4 r3 = *(const float4*)(p + 20);
      v16h a;
      a[0]  = (_Float16)r0.x; a[1]  = (_Float16)r0.y; a[2]  = (_Float16)r0.z; a[3]  = (_Float16)r0.w;
      a[4]  = (_Float16)r1.x; a[5]  = (_Float16)r1.y; a[6]  = (_Float16)r1.z; a[7]  = (_Float16)r1.w;
      a[8]  = (_Float16)r2.x; a[9]  = (_Float16)r2.y; a[10] = (_Float16)r2.z; a[11] = (_Float16)r2.w;
      a[12] = (_Float16)r3.x; a[13] = (_Float16)r3.y; a[14] = (_Float16)r3.z; a[15] = (_Float16)r3.w;
#pragma unroll
      for (int n = 0; n < 8; ++n) {
        const v16h b = *(const v16h*)(smem + LDS_FRAG0 + (n * 8 + kc) * 1024 + lane * 32);
        acc[n] = __builtin_amdgcn_wmma_f32_16x16x32_f16(false, a, false, b,
                                                        (short)0, acc[n], false, false);
      }
    }

    // bias + GELU, spill H1 (f16, row-major 16x128) to per-wave LDS
#pragma unroll
    for (int n = 0; n < 8; ++n) {
#pragma unroll
      for (int v = 0; v < 8; ++v) {
        float x = acc[n][v] + bias0[n];
        h1ws[(M0 + v) * NH1 + n * 16 + colL] = (_Float16)gelu_erf(x);
      }
    }

    // ---- Layer 1: [16 x 128] x [128 x 64] -> [16 x 64] -------------------
    v8f acc2[4];
#pragma unroll
    for (int n = 0; n < 4; ++n) acc2[n] = vzero;

#pragma unroll
    for (int kc = 0; kc < 4; ++kc) {
      const _Float16* hp = h1ws + colL * NH1 + kc * 32 + aoff;
      v8h lo = *(const v8h*)(hp);        // K = base .. base+7
      v8h hi = *(const v8h*)(hp + 16);   // K = base+16 .. base+23
      v16h a;
#pragma unroll
      for (int i = 0; i < 8; ++i) { a[i] = lo[i]; a[8 + i] = hi[i]; }
#pragma unroll
      for (int n = 0; n < 4; ++n) {
        const v16h b = *(const v16h*)(smem + LDS_FRAG1 + (n * 4 + kc) * 1024 + lane * 32);
        acc2[n] = __builtin_amdgcn_wmma_f32_16x16x32_f16(false, a, false, b,
                                                         (short)0, acc2[n], false, false);
      }
    }

#pragma unroll
    for (int n = 0; n < 4; ++n) {
#pragma unroll
      for (int v = 0; v < 8; ++v) {
        float x = acc2[n][v] + bias1[n];
        h2ws[(M0 + v) * NH2 + n * 16 + colL] = (_Float16)gelu_erf(x);
      }
    }

    // ---- Layer 2: [16 x 64] . w2 -> [16 x 1] (split dot over half-lanes) --
    const _Float16* p2 = h2ws + colL * NH2 + hiH * 32;
    float sum = 0.f;
#pragma unroll
    for (int c = 0; c < 4; ++c) {
      v8h hv = *(const v8h*)(p2 + c * 8);
#pragma unroll
      for (int i = 0; i < 8; ++i) sum += (float)hv[i] * w2r[c * 8 + i];
    }
    sum += __shfl_xor(sum, 16, 32);
    if (lane < 16 && myidx >= 0) out[myidx] = sum + b2s;
  }
}

// --- launcher ---------------------------------------------------------------

extern "C" void kernel_launch(void* const* d_in, const int* in_sizes, int n_in,
                              void* d_out, int out_size, void* d_ws, size_t ws_size,
                              hipStream_t stream) {
  const float* X      = (const float*)d_in[0];
  const int*  species = (const int*)d_in[1];
  const float* W0 = (const float*)d_in[2];
  const float* B0 = (const float*)d_in[3];
  const float* W1 = (const float*)d_in[4];
  const float* B1 = (const float*)d_in[5];
  const float* W2 = (const float*)d_in[6];
  const float* B2 = (const float*)d_in[7];
  float* out = (float*)d_out;
  const int N = in_sizes[1];   // N_ATOMS (species element count)

  // workspace layout
  int* hdr     = (int*)d_ws;
  int* counts  = hdr;          // [0..3]
  int* starts  = hdr + 4;      // [4..8]  (5 entries, starts[4] = end)
  int* cursors = hdr + 12;     // [12..15]
  int* idxbuf  = hdr + 16;     // N + 64 ints (16-aligned regions, -1 = pad)
  size_t fragoff = ((size_t)64 + (size_t)(N + 64) * 4 + 255) & ~(size_t)255;
  _Float16* frag0 = (_Float16*)((char*)d_ws + fragoff);           // 256 KB
  _Float16* frag1 = frag0 + (size_t)NTYPE * 8 * 8 * 512;          // +64 KB

  const int total = N + 64;
  k_init   <<<dim3((total + 255) / 256), dim3(256), 0, stream>>>(hdr, idxbuf, total);
  k_count  <<<dim3((N + 255) / 256),     dim3(256), 0, stream>>>(species, N, counts);
  k_starts <<<dim3(1), dim3(1), 0, stream>>>(counts, starts);
  k_scatter<<<dim3((N + 255) / 256),     dim3(256), 0, stream>>>(species, N, starts, cursors, idxbuf);
  k_prep0  <<<dim3(512), dim3(256), 0, stream>>>(W0, frag0);
  k_prep1  <<<dim3(128), dim3(256), 0, stream>>>(W1, frag1);
  k_moe_mlp<<<dim3(4 * BPS), dim3(256), LDS_TOTAL, stream>>>(
      X, idxbuf, starts, frag0, frag1, B0, B1, W2, B2, out);
}
